// DMS_STAttention_67508295959036
// MI455X (gfx1250) — matrix operand order repaired and checked by reading, not compiled, not run on hardware
//
#include <hip/hip_runtime.h>

// ---------- types ----------
typedef unsigned short bfraw;
typedef __bf16 bf16_t;
typedef bf16_t v16bf __attribute__((ext_vector_type(16)));
typedef bfraw  v16u  __attribute__((ext_vector_type(16)));
typedef bfraw  v8u   __attribute__((ext_vector_type(8)));
typedef float  v8f   __attribute__((ext_vector_type(8)));

// native f32->bf16 convert (RNE); lets the backend use v_cvt_pk_bf16_f32
__device__ __forceinline__ bfraw f2bf(float f) {
  return __builtin_bit_cast(bfraw, (bf16_t)f);
}
__device__ __forceinline__ float bf2f(bfraw h) {
  return __uint_as_float(((unsigned)h) << 16);
}
__device__ __forceinline__ float rdf(const float* p) { return *p; }
__device__ __forceinline__ float rdf(const bfraw* p) { return bf2f(*p); }
__device__ __forceinline__ void  wrf(float* p, float v) { *p = v; }
__device__ __forceinline__ void  wrf(bfraw* p, float v) { *p = f2bf(v); }

// ---------- generic WMMA matmul: D[MxN] (+)= A[MxK] * B[KxN] ----------
// A: stored [M x K] bf16, 16B-aligned rows -> 2x ds_load_b128 per 32-K chunk.
//    A-operand half h of lane l: K = (h&7) + ((h>>3)<<4) + 8*(l>=16)
// B: stored TRANSPOSED [N x K] bf16 -> 2x ds_load_b128 per 32-K chunk.
//    B-operand half h of lane l: K = (h&7) + ((h>>3)<<4) + 8*(l>=16)
// TRC: store D transposed ([N x M] bf16) -> one ds_store_b128 per lane.
// ACC: D = A*B + C, C read/written as f32 (fusion accumulator).
// K must be covered by zeros up to round32(K) columns in both operands.
template<bool ACC, bool TRC, typename TC>
__device__ __forceinline__ void mm(const bfraw* __restrict__ A, int ldA,
                                   const bfraw* __restrict__ B, int ldB,
                                   TC* __restrict__ C, int ldC,
                                   int M, int N, int K)
{
  const int lane  = threadIdx.x & 31;
  const int wave  = threadIdx.x >> 5;
  const int nw    = blockDim.x >> 5;
  const int rl    = lane & 15;
  const int hiA   = (lane >> 4) << 3;   // +8  K-offset (A)
  const int hiB   = (lane >> 4) << 3;   // +8  K-offset (B)
  const int rowHi = (lane >> 4) << 3;   // +8  M-offset (C/D)
  const int tN    = N >> 4;
  const int tiles = (M >> 4) * tN;

  for (int t = wave; t < tiles; t += nw) {
    const int ti = t / tN;
    const int tm = ti << 4;
    const int tn = (t - ti * tN) << 4;
    v8f acc;
#pragma unroll
    for (int r = 0; r < 8; ++r)
      acc[r] = ACC ? rdf(&C[(tm + r + rowHi) * ldC + tn + rl]) : 0.0f;

    for (int kk = 0; kk < K; kk += 32) {
      const v8u a0 = *(const v8u*)(A + (tm + rl) * ldA + kk + hiA);
      const v8u a1 = *(const v8u*)(A + (tm + rl) * ldA + kk + hiA + 16);
      const v8u b0 = *(const v8u*)(B + (tn + rl) * ldB + kk + hiB);
      const v8u b1 = *(const v8u*)(B + (tn + rl) * ldB + kk + hiB + 16);
      const v16u av = __builtin_shufflevector(a0, a1, 0,1,2,3,4,5,6,7,8,9,10,11,12,13,14,15);
      const v16u bv = __builtin_shufflevector(b0, b1, 0,1,2,3,4,5,6,7,8,9,10,11,12,13,14,15);
      acc = __builtin_amdgcn_wmma_f32_16x16x32_bf16(
                false, __builtin_bit_cast(v16bf, av),
                false, __builtin_bit_cast(v16bf, bv),
                (short)0, acc, false, false);
    }

    if constexpr (TRC) {
      v8u cv;
#pragma unroll
      for (int r = 0; r < 8; ++r) cv[r] = f2bf(acc[r]);
      *(v8u*)(C + (tn + rl) * ldC + tm + rowHi) = cv;
    } else {
#pragma unroll
      for (int r = 0; r < 8; ++r)
        wrf(&C[(tm + r + rowHi) * ldC + tn + rl], acc[r]);
    }
  }
}

// ---------- GAT additive scores (x stored transposed: [C x nodes]) ----------
template<int NODES>
__device__ __forceinline__ void scores(const bfraw* xT, int ldx,
                                       const float* wsrc, const float* wdst,
                                       float* sv, float* tv, int nvalid)
{
  for (int i = threadIdx.x; i < NODES; i += blockDim.x) {
    float s = 0.f, t = 0.f;
    if (i < nvalid) {
#pragma unroll
      for (int c = 0; c < 64; ++c) {
        const float xv = bf2f(xT[c * ldx + i]);
        s = fmaf(xv, wsrc[c], s);
        t = fmaf(xv, wdst[c], t);
      }
    }
    sv[i] = s; tv[i] = t;
  }
}

// ---------- attn[i,j] = leaky_relu(rv[i] + cv[j]); zero outside valid ----------
__device__ __forceinline__ void attn_fill(bfraw* A, int ldA, const float* rv, const float* cv,
                                          int NR, int NC, int nvalid)
{
  for (int idx = threadIdx.x; idx < NR * NC; idx += blockDim.x) {
    const int i = idx / NC, j = idx - i * NC;
    float v = 0.f;
    if (i < nvalid && j < nvalid) {
      const float e = rv[i] + cv[j];
      v = e > 0.f ? e : 0.2f * e;
    }
    A[i * ldA + j] = f2bf(v);
  }
}

// ---------- scale-0 fill: writes f32 fusion accumulator AND bf16 operand copy ----------
__device__ __forceinline__ void attn_fill0(float* Fb, int ldF, bfraw* A0, int ldA0,
                                           const float* rv, const float* cv,
                                           int NR, int nvalid)
{
  for (int idx = threadIdx.x; idx < NR * NR; idx += blockDim.x) {
    const int i = idx / NR, j = idx - i * NR;
    float v = 0.f;
    if (i < nvalid && j < nvalid) {
      const float e = rv[i] + cv[j];
      v = e > 0.f ? e : 0.2f * e;
    }
    Fb[i * ldF + j] = v;
    A0[i * ldA0 + j] = f2bf(v);
  }
}

// ---------- row softmax with bias; zero K-pads; optional transposed copy ----------
template<int NC, int KP, bool WRT>
__device__ __forceinline__ void softmax_rows(bfraw* S, int ldS, const float* bias,
                                             bfraw* ST, int ldST,
                                             int nrowpad, int nrow, int ncol)
{
  for (int i = threadIdx.x; i < nrowpad; i += blockDim.x) {
    float v[NC];
    const bool rowok = (i < nrow);
    float m = -1e30f;
#pragma unroll
    for (int k = 0; k < NC; ++k) {
      const float val = (rowok && k < ncol) ? (bf2f(S[i * ldS + k]) + bias[k]) : -1e30f;
      v[k] = val; m = fmaxf(m, val);
    }
    float sum = 0.f;
#pragma unroll
    for (int k = 0; k < NC; ++k) {
      const float e = (k < ncol) ? __expf(v[k] - m) : 0.f;
      v[k] = e; sum += e;
    }
    const float inv = rowok ? (1.f / sum) : 0.f;
#pragma unroll
    for (int k = 0; k < NC; ++k) {
      const bfraw o = f2bf(v[k] * inv);
      S[i * ldS + k] = o;
      if constexpr (WRT) ST[k * ldST + i] = o;
    }
#pragma unroll
    for (int k = NC; k < KP; ++k) S[i * ldS + k] = 0;   // K-pad to 32
  }
}

// ---------- softmax on a transposed-stored matrix ----------
template<int NCOLS, int KP>
__device__ __forceinline__ void softmax_rows_T(bfraw* ST, int ldST, const float* bias,
                                               int nrow, int ncol)
{
  for (int i = threadIdx.x; i < KP; i += blockDim.x) {   // logical rows incl. K-pad
    float v[NCOLS];
    const bool rowok = (i < nrow);
    float m = -1e30f;
#pragma unroll
    for (int k = 0; k < NCOLS; ++k) {
      const float val = (rowok && k < ncol) ? (bf2f(ST[k * ldST + i]) + bias[k]) : -1e30f;
      v[k] = val; m = fmaxf(m, val);
    }
    float sum = 0.f;
#pragma unroll
    for (int k = 0; k < NCOLS; ++k) {
      const float e = (k < ncol) ? __expf(v[k] - m) : 0.f;
      v[k] = e; sum += e;
    }
    const float inv = rowok ? (1.f / sum) : 0.f;
#pragma unroll
    for (int k = 0; k < NCOLS; ++k)
      ST[k * ldST + i] = f2bf(v[k] * inv);
  }
}

__device__ __forceinline__ void zero_buf(bfraw* p, int n) {
  for (int i = threadIdx.x; i < n; i += blockDim.x) p[i] = 0;
}

// ---------- precompute wa = W @ a for 12 attention vectors ----------
__global__ void wa_kernel(const float* __restrict__ W,
                          const float* a0, const float* a1, const float* a2,
                          const float* a3, const float* a4, const float* a5,
                          const float* a6, const float* a7, const float* a8,
                          const float* a9, const float* a10, const float* a11,
                          float* __restrict__ wa)
{
  const float* as[12] = {a0,a1,a2,a3,a4,a5,a6,a7,a8,a9,a10,a11};
  const float* a = as[blockIdx.x];
  const int c = threadIdx.x;              // 64 threads
  float s = 0.f;
  for (int h = 0; h < 128; ++h) s = fmaf(W[c * 128 + h], a[h], s);
  wa[blockIdx.x * 64 + c] = s;
}

// ---------- one full multi-scale GAT branch, one graph per workgroup ----------
template<int n0, int N0, int n1, int N1, int n2, int N2, bool SPATIAL>
__global__ __launch_bounds__(256)
void branch_kernel(const float* __restrict__ src,   // [B, 64, T, J]
                   const float* __restrict__ wa,    // 6 x 64 (s0,d0,s1,d1,s2,d2)
                   const float* __restrict__ Wp0g, const float* __restrict__ bp0,
                   const float* __restrict__ Wp1g, const float* __restrict__ bp1,
                   float* __restrict__ out,         // [G, n0, n0]
                   int B, int T, int J)
{
  constexpr int C    = 64;
  constexpr int LD64 = 72;                 // ld for 64-col bf16 buffers
  constexpr int LDN0 = N0 + 8;             // x0T / attn0b cols
  constexpr int LDP  = 40;                 // ld for <=32-col bf16 buffers (K-pad 32)
  constexpr int LDF  = N0 + 1;             // f32 fusion accumulator

  // region sizes (elements) with lifetime aliasing
  constexpr int XBE = C * LDN0;                                    // x0T -> attn1/attn1T/attn2T
  constexpr int ABE = (N0 * LDN0 > C * LDP) ? N0 * LDN0 : C * LDP; // attn0b -> x1T
  constexpr int YE  = (N0 * LD64 > C * LDP) ? N0 * LD64 : C * LDP; // y0/S0T/y1 -> x2T
  constexpr int U1E = (N1 * LD64 > N0 * LDP) ? N1 * LD64 : N0 * LDP; // Wp0T -> T1/T2
  constexpr int U2E = (N2 * LD64 > N0 * LDP) ? N2 * LD64 : N0 * LDP; // Wp1T -> S01

  __shared__ __align__(16) bfraw xbuf[XBE];
  __shared__ __align__(16) bfraw ab[ABE];
  __shared__ __align__(16) bfraw y[YE];
  __shared__ __align__(16) bfraw S0[N0 * LDP];
  __shared__ __align__(16) bfraw S1T[N2 * LDP];
  __shared__ __align__(16) bfraw u1[U1E];
  __shared__ __align__(16) bfraw u2[U2E];
  __shared__ float F[N0 * LDF];
  __shared__ float sv[N0], tv[N0];
  __shared__ float wal[6 * 64];

  bfraw* x0T    = xbuf;                  // [C x LDN0]
  bfraw* attn1  = xbuf;                  // [N1 x LDP]  (x0T dead)
  bfraw* attn1T = xbuf + N1 * LDP;       // [N1 x LDP]
  bfraw* attn2T = xbuf + 2 * N1 * LDP;   // [N2 x LDP]
  bfraw* attn0b = ab;                    // [N0 x LDN0]
  bfraw* x1T    = ab;                    // [C x LDP]   (attn0b dead)
  bfraw* x2T    = y;                     // [C x LDP]   (y1 dead)

  const int g   = blockIdx.x;
  const int tid = threadIdx.x;
  const int bd  = blockDim.x;

  int b, fix;
  if (SPATIAL) { b = g / T; fix = g - b * T; }   // t fixed, nodes = joints (J)
  else         { b = g / J; fix = g - b * J; }   // j fixed, nodes = time (T)

  // ---- loads: x0^T bf16 (zero node-pads), Wp^T, wa table ----
  for (int idx = tid; idx < C * N0; idx += bd) {
    const int c = idx / N0, node = idx - c * N0;
    float v = 0.f;
    if (node < n0) {
      const size_t off = SPATIAL
        ? (((size_t)(b * C + c) * T + fix)  * J + node)
        : (((size_t)(b * C + c) * T + node) * J + fix);
      v = src[off];
    }
    x0T[c * LDN0 + node] = f2bf(v);
  }
  for (int idx = tid; idx < C * N1; idx += bd) {
    const int c = idx / N1, k = idx - c * N1;
    u1[k * LD64 + c] = (k < n1) ? f2bf(Wp0g[c * n1 + k]) : (bfraw)0;   // Wp0^T
  }
  for (int idx = tid; idx < C * N2; idx += bd) {
    const int c = idx / N2, k = idx - c * N2;
    u2[k * LD64 + c] = (k < n2) ? f2bf(Wp1g[c * n2 + k]) : (bfraw)0;   // Wp1^T
  }
  for (int idx = tid; idx < 6 * 64; idx += bd) wal[idx] = wa[idx];
  __syncthreads();

  // ---- scale 0 ----
  scores<N0>(x0T, LDN0, wal + 0 * 64, wal + 1 * 64, sv, tv, n0);
  __syncthreads();
  attn_fill0(F, LDF, attn0b, LDN0, sv, tv, N0, n0);       // F (f32) + attn0 (bf16)
  __syncthreads();
  // y0 = attn0 @ x0 : flipped (y0^T = x0^T @ attn0^T), TRC -> y0 normal in y
  mm<false,true>(x0T, LDN0, attn0b, LDN0, y, LD64, C, N0, N0);
  __syncthreads();
  // logits0: flipped (S0^T = Wp0^T @ y0^T), TRC -> S0 normal; clear x1T (attn0b dead)
  mm<false,true>(u1, LD64, y, LD64, S0, LDP, N1, N0, C);
  zero_buf(x1T, C * LDP);
  __syncthreads();
  softmax_rows<N1,32,true>(S0, LDP, bp0, y, LD64, N0, n0, n1);  // S0 (+ S0^T into y)
  __syncthreads();
  // x1 = S0^T @ x0 : TRC -> x1^T
  mm<false,true>(y, LD64, x0T, LDN0, x1T, LDP, N1, C, N0);
  __syncthreads();

  // ---- scale 1 ----
  scores<N1>(x1T, LDP, wal + 2 * 64, wal + 3 * 64, sv, tv, n1);
  __syncthreads();
  attn_fill(attn1,  LDP, sv, tv, N1, 32, n1);             // xbuf region (x0T dead)
  attn_fill(attn1T, LDP, tv, sv, N1, 32, n1);
  __syncthreads();
  // y1 = attn1 @ x1 : flipped (y1^T = x1^T @ attn1^T), TRC -> y1 normal in y
  mm<false,true>(x1T, LDP, attn1, LDP, y, LD64, C, N1, N1);
  __syncthreads();
  // logits1 = y1 @ Wp1 : TRC -> S1^T
  mm<false,true>(y, LD64, u2, LD64, S1T, LDP, N1, N2, C);
  __syncthreads();
  softmax_rows_T<N2,32>(S1T, LDP, bp1, n1, n2);
  __syncthreads();
  // x2 = S1^T @ x1 : TRC -> x2^T in y (y1 dead)
  mm<false,true>(S1T, LDP, x1T, LDP, x2T, LDP, N2, C, N1);
  __syncthreads();

  // ---- scale 2 ----
  scores<N2>(x2T, LDP, wal + 4 * 64, wal + 5 * 64, sv, tv, n2);
  __syncthreads();
  attn_fill(attn2T, LDP, tv, sv, N2, 32, n2);
  zero_buf(u1, N0 * LDP);                                 // T1 (Wp0^T dead)
  zero_buf(u2, N0 * LDP);                                 // S01 (Wp1^T dead)
  __syncthreads();

  // ---- fusion: F = attn0 + S0 attn1 S0^T + (S0 S1) attn2 (S0 S1)^T ----
  // T1 = S0 @ attn1 : flipped (T1^T = attn1^T @ S0^T), TRC -> T1 normal in u1
  mm<false,true>(attn1T, LDP, S0, LDP, u1, LDP, N1, N0, N1);
  __syncthreads();
  mm<true,false>(u1, LDP, S0, LDP, F, LDF, N0, N0, N1);   // F += T1 @ S0^T
  __syncthreads();
  // S01 = S0 @ S1 : flipped (S01^T = S1^T @ S0^T), TRC -> S01 normal in u2
  mm<false,true>(S1T, LDP, S0, LDP, u2, LDP, N2, N0, N1);
  __syncthreads();
  zero_buf(u1, N0 * LDP);                                 // clear for T2 K-pads
  __syncthreads();
  // T2 = S01 @ attn2 : flipped (T2^T = attn2^T @ S01^T), TRC -> T2 normal in u1
  mm<false,true>(attn2T, LDP, u2, LDP, u1, LDP, N2, N0, N2);
  __syncthreads();
  mm<true,false>(u1, LDP, u2, LDP, F, LDF, N0, N0, N2);   // F += T2 @ S01^T
  __syncthreads();

  // ---- final row softmax, write out ----
  float* og = out + (size_t)g * n0 * n0;
  for (int i = tid; i < n0; i += bd) {
    float m = -1e30f;
    for (int j = 0; j < n0; ++j) m = fmaxf(m, F[i * LDF + j]);
    float s = 0.f;
    for (int j = 0; j < n0; ++j) s += __expf(F[i * LDF + j] - m);
    const float inv = 1.f / s;
    for (int j = 0; j < n0; ++j) og[i * n0 + j] = __expf(F[i * LDF + j] - m) * inv;
  }
}

// ---------- launch ----------
extern "C" void kernel_launch(void* const* d_in, const int* in_sizes, int n_in,
                              void* d_out, int out_size, void* d_ws, size_t ws_size,
                              hipStream_t stream)
{
  (void)in_sizes; (void)n_in; (void)out_size; (void)ws_size;
  const float* src  = (const float*)d_in[0];
  const float* W    = (const float*)d_in[1];
  const float* Wsp0 = (const float*)d_in[14];
  const float* bsp0 = (const float*)d_in[15];
  const float* Wsp1 = (const float*)d_in[16];
  const float* bsp1 = (const float*)d_in[17];
  const float* Wtp0 = (const float*)d_in[18];
  const float* btp0 = (const float*)d_in[19];
  const float* Wtp1 = (const float*)d_in[20];
  const float* btp1 = (const float*)d_in[21];

  float* wa  = (float*)d_ws;             // 12 x 64 floats
  float* out = (float*)d_out;

  const int B = 256, T = 64, J = 25;

  // wa slots: [0..5] spatial (s0,d0,s1,d1,s2,d2), [6..11] temporal
  wa_kernel<<<12, 64, 0, stream>>>(W,
      (const float*)d_in[2],  (const float*)d_in[3],    // as_src0, as_dst0
      (const float*)d_in[6],  (const float*)d_in[7],    // as_src1, as_dst1
      (const float*)d_in[10], (const float*)d_in[11],   // as_src2, as_dst2
      (const float*)d_in[4],  (const float*)d_in[5],    // at_src0, at_dst0
      (const float*)d_in[8],  (const float*)d_in[9],    // at_src1, at_dst1
      (const float*)d_in[12], (const float*)d_in[13],   // at_src2, at_dst2
      wa);

  // spatial: 16384 graphs of 25 nodes (pad 32/16/16)
  branch_kernel<25,32,12,16,5,16,true><<<B * T, 256, 0, stream>>>(
      src, wa, Wsp0, bsp0, Wsp1, bsp1, out, B, T, J);

  // temporal: 6400 graphs of 64 nodes (64/32/16)
  branch_kernel<64,64,32,32,16,16,false><<<B * J, 256, 0, stream>>>(
      src, wa + 6 * 64, Wtp0, btp0, Wtp1, btp1,
      out + (size_t)B * T * J * J, B, T, J);
}
